// AttentionHead_42932493091408
// MI455X (gfx1250) — matrix operand order repaired and checked by reading, not compiled, not run on hardware
//
#include <hip/hip_runtime.h>
#include <hip/hip_bf16.h>

// Problem constants (from reference)
constexpr int LSEQ   = 4096;
constexpr int DIM    = 1024;
constexpr int HDIM   = 128;
constexpr int NSPLIT = 4;      // key-range splits per query tile (occupancy)

typedef __attribute__((ext_vector_type(16))) __bf16 v16bf;
typedef __attribute__((ext_vector_type(8)))  __bf16 v8bf;
typedef __attribute__((ext_vector_type(8)))  float  v8f;
typedef __attribute__((ext_vector_type(8)))  float  f32x8;

// ---------------------------------------------------------------------------
// WMMA wrapper: D = A(16x32 bf16) * B(32x16 bf16) + C(16x16 f32)
// ---------------------------------------------------------------------------
__device__ __forceinline__ v8f wmma_bf16(v16bf a, v16bf b, v8f c) {
    return __builtin_amdgcn_wmma_f32_16x16x32_bf16(
        /*neg_a=*/false, a, /*neg_b=*/false, b,
        /*c_mod=*/(short)0, c, /*reuse_a=*/false, /*reuse_b=*/false);
}

// ---------------------------------------------------------------------------
// A-fragment (16x32, bf16): lane = M (row), element e<8 -> K=hs*8+e,
// e>=8 -> K=16+hs*8+(e-8), where hs = lane>>4. Source: row-major bf16.
// ---------------------------------------------------------------------------
__device__ __forceinline__ v16bf load_a_bf16(const __bf16* __restrict__ p,
                                             int ld, int row0, int k0) {
    int lane = threadIdx.x & 31;
    int m = lane & 15, hs = lane >> 4;
    const __bf16* base = p + (size_t)(row0 + m) * ld + k0 + hs * 8;
    v8bf lo = *(const v8bf*)(base);
    v8bf hi = *(const v8bf*)(base + 16);
    v16bf r;
#pragma unroll
    for (int e = 0; e < 8; ++e) { r[e] = lo[e]; r[8 + e] = hi[e]; }
    return r;
}

// Same A-fragment but from f32 row-major source, converting to bf16.
__device__ __forceinline__ v16bf load_a_f32(const float* __restrict__ p,
                                            int ld, int row0, int k0) {
    int lane = threadIdx.x & 31;
    int m = lane & 15, hs = lane >> 4;
    const float* base = p + (size_t)(row0 + m) * ld + k0 + hs * 8;
    f32x8 lo = *(const f32x8*)(base);
    f32x8 hi = *(const f32x8*)(base + 16);
    v16bf r;
#pragma unroll
    for (int e = 0; e < 8; ++e) {
        r[e]     = (__bf16)lo[e];
        r[8 + e] = (__bf16)hi[e];
    }
    return r;
}

// ---------------------------------------------------------------------------
// B-fragment (32x16, bf16): lane = N (column), element e<8 -> K=hs*8+e,
// e>=8 -> K=16+hs*8+(e-8), where hs = lane>>4.
// Source laid out "N-major": row n of `p` holds B[:,n] contiguously in K.
// ---------------------------------------------------------------------------
__device__ __forceinline__ v16bf load_b_bf16(const __bf16* __restrict__ p,
                                             int ld, int n0, int k0) {
    int lane = threadIdx.x & 31;
    int n = lane & 15, hs = lane >> 4;
    const __bf16* base = p + (size_t)(n0 + n) * ld + k0 + hs * 8;
    v8bf lo = *(const v8bf*)(base);
    v8bf hi = *(const v8bf*)(base + 16);
    v16bf r;
#pragma unroll
    for (int e = 0; e < 8; ++e) { r[e] = lo[e]; r[8 + e] = hi[e]; }
    return r;
}

// ---------------------------------------------------------------------------
// Kernel 1: convert + transpose weights: wt[mat][h][d] = bf16(W_mat[d][h])
// ---------------------------------------------------------------------------
__global__ void prep_weights_kernel(const float* __restrict__ Wq,
                                    const float* __restrict__ Wk,
                                    const float* __restrict__ Wv,
                                    __bf16* __restrict__ wt) {
    int idx = blockIdx.x * blockDim.x + threadIdx.x;
    int total = 3 * HDIM * DIM;
    if (idx >= total) return;
    int mat = idx / (HDIM * DIM);
    int rem = idx - mat * (HDIM * DIM);
    int h = rem / DIM;
    int d = rem - h * DIM;
    const float* W = (mat == 0) ? Wq : (mat == 1 ? Wk : Wv);
    wt[idx] = (__bf16)W[(size_t)d * HDIM + h];
}

// ---------------------------------------------------------------------------
// Kernel 2: q/k/v = x @ W + b. One wave per (16-row tile, matrix).
// q,k stored row-major bf16 [L][H]; v stored transposed bf16 [H][L].
// B-fragments staged into an array before the WMMA chain so the scheduler
// gets a full clause of independent b128 loads to hide under WMMAs.
// ---------------------------------------------------------------------------
__global__ void __launch_bounds__(32)
qkv_kernel(const float* __restrict__ x,
           const float* __restrict__ bq, const float* __restrict__ bk,
           const float* __restrict__ bv,
           const __bf16* __restrict__ wt,   // [3][H][D]
           __bf16* __restrict__ qbf, __bf16* __restrict__ kbf,
           __bf16* __restrict__ vt) {
    int mat = blockIdx.y;
    const __bf16* w = wt + (size_t)mat * HDIM * DIM;
    const float* bias = (mat == 0) ? bq : (mat == 1 ? bk : bv);
    int i0 = blockIdx.x * 16;
    int lane = threadIdx.x & 31;
    int nlo = lane & 15, hs = lane >> 4;

    v8f acc[8];
#pragma unroll
    for (int nt = 0; nt < 8; ++nt) acc[nt] = (v8f)(0.0f);

    for (int k0 = 0; k0 < DIM; k0 += 32) {
        v16bf a = load_a_f32(x, DIM, i0, k0);
        v16bf bfr[8];
#pragma unroll
        for (int nt = 0; nt < 8; ++nt) bfr[nt] = load_b_bf16(w, DIM, nt * 16, k0);
#pragma unroll
        for (int nt = 0; nt < 8; ++nt) acc[nt] = wmma_bf16(a, bfr[nt], acc[nt]);
    }

#pragma unroll
    for (int nt = 0; nt < 8; ++nt) {
        int col = nt * 16 + nlo;
        float bb = bias[col];
        if (mat < 2) {
            __bf16* dst = (mat == 0) ? qbf : kbf;
#pragma unroll
            for (int r = 0; r < 8; ++r) {
                int row = i0 + hs * 8 + r;
                dst[(size_t)row * HDIM + col] = (__bf16)(acc[nt][r] + bb);
            }
        } else {
            v8bf pk;
#pragma unroll
            for (int r = 0; r < 8; ++r) pk[r] = (__bf16)(acc[nt][r] + bb);
            *(v8bf*)(vt + (size_t)col * LSEQ + i0 + hs * 8) = pk;
        }
    }
}

// ---------------------------------------------------------------------------
// Kernel 3: split-K flash attention partials. grid = (L/16, NSPLIT), one wave
// each. Split `s` handles key-block range [nb*s/NSPLIT, nb*(s+1)/NSPLIT) of
// the causal range for its query tile; emits unnormalized (O, m, l).
// ---------------------------------------------------------------------------
__global__ void __launch_bounds__(32)
attn_partial_kernel(const __bf16* __restrict__ qbf,
                    const __bf16* __restrict__ kbf,
                    const __bf16* __restrict__ vt,
                    float* __restrict__ opart,   // [NSPLIT][L][H]
                    float* __restrict__ mpart,   // [NSPLIT][L]
                    float* __restrict__ lpart) { // [NSPLIT][L]
    __shared__ float ptile[16][33];   // pad to kill 2-way bank conflicts

    int i0 = blockIdx.x * 16;
    int split = blockIdx.y;
    int lane = threadIdx.x & 31;
    int nlo = lane & 15, hs = lane >> 4;
    const float scale = 0.08838834764831843f;  // 1/sqrt(128)

    int nb = (i0 + 15) / 32 + 1;               // causal 32-key blocks
    int b0 = (nb * split) / NSPLIT;
    int b1 = (nb * (split + 1)) / NSPLIT;

    v16bf aq[4];
#pragma unroll
    for (int t = 0; t < 4; ++t) aq[t] = load_a_bf16(qbf, HDIM, i0, t * 32);

    v8f o[8];
#pragma unroll
    for (int nt = 0; nt < 8; ++nt) o[nt] = (v8f)(0.0f);
    float m_r[8], l_r[8];
#pragma unroll
    for (int r = 0; r < 8; ++r) { m_r[r] = -1.0e30f; l_r[r] = 0.0f; }

    for (int b = b0; b < b1; ++b) {
        int j0 = b * 32;

        // prefetch next key block's K rows (global_prefetch_b8)
        if (b + 1 < b1) {
            const __bf16* pf = kbf + (size_t)(j0 + 32 + lane) * HDIM;
            __builtin_prefetch(pf, 0, 0);
            __builtin_prefetch(pf + 64, 0, 0);
        }

        // ---- stage all 8 K B-fragments, then the WMMA chains ----
        v16bf bk[8];
#pragma unroll
        for (int t = 0; t < 4; ++t) {
            bk[t]     = load_b_bf16(kbf, HDIM, j0,      t * 32);
            bk[4 + t] = load_b_bf16(kbf, HDIM, j0 + 16, t * 32);
        }
        v8f s0 = (v8f)(0.0f), s1 = (v8f)(0.0f);
#pragma unroll
        for (int t = 0; t < 4; ++t) s0 = wmma_bf16(aq[t], bk[t],     s0);
#pragma unroll
        for (int t = 0; t < 4; ++t) s1 = wmma_bf16(aq[t], bk[4 + t], s1);

        // ---- scale + causal mask ----
#pragma unroll
        for (int r = 0; r < 8; ++r) {
            int row = i0 + hs * 8 + r;
            float u0 = s0[r] * scale;
            float u1 = s1[r] * scale;
            if (j0 + nlo > row)      u0 = -1.0e30f;
            if (j0 + 16 + nlo > row) u1 = -1.0e30f;
            s0[r] = u0; s1[r] = u1;
        }

        // ---- online softmax (rows 0-7 in lanes 0-15, rows 8-15 in 16-31) --
        float rscale[8];
#pragma unroll
        for (int r = 0; r < 8; ++r) {
            float mx = fmaxf(s0[r], s1[r]);
            mx = fmaxf(mx, __shfl_xor(mx, 1));
            mx = fmaxf(mx, __shfl_xor(mx, 2));
            mx = fmaxf(mx, __shfl_xor(mx, 4));
            mx = fmaxf(mx, __shfl_xor(mx, 8));
            float mnew = fmaxf(m_r[r], mx);
            float sc = __expf(m_r[r] - mnew);
            float p0 = __expf(s0[r] - mnew);
            float p1 = __expf(s1[r] - mnew);
            float rs = p0 + p1;
            rs += __shfl_xor(rs, 1);
            rs += __shfl_xor(rs, 2);
            rs += __shfl_xor(rs, 4);
            rs += __shfl_xor(rs, 8);
            l_r[r] = l_r[r] * sc + rs;
            m_r[r] = mnew;
            rscale[r] = sc;
            ptile[hs * 8 + r][nlo]      = p0;  // D-layout -> LDS
            ptile[hs * 8 + r][16 + nlo] = p1;
        }

        // ---- rescale accumulators ----
#pragma unroll
        for (int nt = 0; nt < 8; ++nt)
#pragma unroll
            for (int r = 0; r < 8; ++r) o[nt][r] *= rscale[r];

        // ---- reread P in A-fragment layout (single wave: LDS in-order) ----
        v16bf ap;
#pragma unroll
        for (int e = 0; e < 8; ++e) {
            ap[e]     = (__bf16)ptile[nlo][hs * 8 + e];
            ap[8 + e] = (__bf16)ptile[nlo][16 + hs * 8 + e];
        }

        // ---- O += P * V : stage 8 V B-fragments, then WMMA chain ----
        v16bf bvf[8];
#pragma unroll
        for (int nt = 0; nt < 8; ++nt) bvf[nt] = load_b_bf16(vt, LSEQ, nt * 16, j0);
#pragma unroll
        for (int nt = 0; nt < 8; ++nt) o[nt] = wmma_bf16(ap, bvf[nt], o[nt]);
    }

    // ---- store unnormalized partials ----
#pragma unroll
    for (int nt = 0; nt < 8; ++nt) {
        int col = nt * 16 + nlo;
#pragma unroll
        for (int r = 0; r < 8; ++r) {
            int row = i0 + hs * 8 + r;
            opart[((size_t)split * LSEQ + row) * HDIM + col] = o[nt][r];
        }
    }
    if (nlo == 0) {   // lanes 0 and 16 cover rows 0-7 / 8-15
#pragma unroll
        for (int r = 0; r < 8; ++r) {
            int row = i0 + hs * 8 + r;
            mpart[(size_t)split * LSEQ + row] = m_r[r];
            lpart[(size_t)split * LSEQ + row] = l_r[r];
        }
    }
}

// ---------------------------------------------------------------------------
// Kernel 4: combine split-K partials.
// out[row,col] = sum_s e^{m_s-m*} O_s / sum_s e^{m_s-m*} l_s
// ---------------------------------------------------------------------------
__global__ void combine_kernel(const float* __restrict__ opart,
                               const float* __restrict__ mpart,
                               const float* __restrict__ lpart,
                               float* __restrict__ out) {
    int idx = blockIdx.x * blockDim.x + threadIdx.x;
    if (idx >= LSEQ * HDIM) return;
    int row = idx / HDIM;
    int col = idx - row * HDIM;

    float mmax = -1.0e30f;
#pragma unroll
    for (int s = 0; s < NSPLIT; ++s)
        mmax = fmaxf(mmax, mpart[(size_t)s * LSEQ + row]);

    float lsum = 0.0f, acc = 0.0f;
#pragma unroll
    for (int s = 0; s < NSPLIT; ++s) {
        float w = __expf(mpart[(size_t)s * LSEQ + row] - mmax);
        lsum += lpart[(size_t)s * LSEQ + row] * w;
        acc  += opart[((size_t)s * LSEQ + row) * HDIM + col] * w;
    }
    out[idx] = acc / lsum;
}

// ---------------------------------------------------------------------------
extern "C" void kernel_launch(void* const* d_in, const int* in_sizes, int n_in,
                              void* d_out, int out_size, void* d_ws, size_t ws_size,
                              hipStream_t stream) {
    const float* x  = (const float*)d_in[0];
    const float* Wq = (const float*)d_in[1];
    const float* bq = (const float*)d_in[2];
    const float* Wk = (const float*)d_in[3];
    const float* bk = (const float*)d_in[4];
    const float* Wv = (const float*)d_in[5];
    const float* bv = (const float*)d_in[6];
    float* out = (float*)d_out;

    char* ws = (char*)d_ws;
    __bf16* wt  = (__bf16*)ws;                                    // 3*H*D bf16
    __bf16* qbf = (__bf16*)(ws + (size_t)3 * HDIM * DIM * 2);     // L*H bf16
    __bf16* kbf = qbf + (size_t)LSEQ * HDIM;                      // L*H bf16
    __bf16* vt  = kbf + (size_t)LSEQ * HDIM;                      // H*L bf16 (transposed)
    float* opart = (float*)(vt + (size_t)LSEQ * HDIM);            // NSPLIT*L*H f32
    float* mpart = opart + (size_t)NSPLIT * LSEQ * HDIM;          // NSPLIT*L f32
    float* lpart = mpart + (size_t)NSPLIT * LSEQ;                 // NSPLIT*L f32

    int prep_total = 3 * HDIM * DIM;
    prep_weights_kernel<<<(prep_total + 255) / 256, 256, 0, stream>>>(Wq, Wk, Wv, wt);

    qkv_kernel<<<dim3(LSEQ / 16, 3), 32, 0, stream>>>(x, bq, bk, bv, wt, qbf, kbf, vt);

    attn_partial_kernel<<<dim3(LSEQ / 16, NSPLIT), 32, 0, stream>>>(
        qbf, kbf, vt, opart, mpart, lpart);

    combine_kernel<<<(LSEQ * HDIM + 255) / 256, 256, 0, stream>>>(
        opart, mpart, lpart, out);
}
